// WindowAttention_40604620816719
// MI455X (gfx1250) — matrix operand order repaired and checked by reading, not compiled, not run on hardware
//
#include <hip/hip_runtime.h>
#include <hip/hip_bf16.h>
#include <math.h>

// ---------------- WMMA types & helpers (gfx1250, wave32) ----------------
typedef __attribute__((ext_vector_type(16))) __bf16 bf16x16;
typedef __attribute__((ext_vector_type(8)))  float  f32x8;

union FragBF { bf16x16 v; unsigned short u[16]; };

__device__ __forceinline__ unsigned short f2bfu(float f) {  // f32 -> bf16, native cvt
  union { __bf16 b; unsigned short u; } x;
  x.b = (__bf16)f;
  return x.u;
}
__device__ __forceinline__ float bfu2f(unsigned short s) {
  union { unsigned u; float f; } a; a.u = ((unsigned)s) << 16;
  return a.f;
}

// A-matrix 16x32 bf16 K order (matches B loaders):
// lanes 0-15: K = {0..15}; lanes 16-31: K = {16..31}
// (two contiguous 16B chunks per lane -> compiler merges to 2x ds_load_b128)
__device__ __forceinline__ int kmapA(int e, int kh) {
  return 16 * kh + e;
}

__device__ __forceinline__ f32x8 wmma_bf16(bf16x16 a, bf16x16 b, f32x8 c) {
  return __builtin_amdgcn_wmma_f32_16x16x32_bf16(false, a, false, b, (short)0, c,
                                                 false, false);
}

// A fragment from row-major [row][k] storage (stride in ushorts)
__device__ __forceinline__ bf16x16 load_fragA(const unsigned short* base, int row0,
                                              int stride, int k0, int lane) {
  FragBF f;
  int m = row0 + (lane & 15);
  int kh = lane >> 4;
#pragma unroll
  for (int e = 0; e < 16; ++e) f.u[e] = base[m * stride + k0 + kmapA(e, kh)];
  return f.v;
}
// B fragment; storage is [n][k] (i.e. B^T row-major). B layout: n=lane%16, k=16*(lane/16)+e
__device__ __forceinline__ bf16x16 load_fragB_nk(const unsigned short* base, int n0,
                                                 int stride, int k0, int lane) {
  FragBF f;
  int n = n0 + (lane & 15);
  int kb = k0 + 16 * (lane >> 4);
#pragma unroll
  for (int e = 0; e < 16; ++e) f.u[e] = base[n * stride + kb + e];
  return f.v;
}
// B fragment; storage is [k][n] row-major
__device__ __forceinline__ bf16x16 load_fragB_kn(const unsigned short* base, int k0,
                                                 int stride, int n0, int lane) {
  FragBF f;
  int n = n0 + (lane & 15);
  int kb = k0 + 16 * (lane >> 4);
#pragma unroll
  for (int e = 0; e < 16; ++e) f.u[e] = base[(kb + e) * stride + n];
  return f.v;
}
// B fragment converted on the fly from f32 global weights W[n][k]
// (16 consecutive f32 per lane -> 4x global_load_b128, GL2-resident weights)
__device__ __forceinline__ bf16x16 load_fragB_w32(const float* base, int n0,
                                                  int stride, int k0, int lane) {
  FragBF f;
  int n = n0 + (lane & 15);
  int kb = k0 + 16 * (lane >> 4);
#pragma unroll
  for (int e = 0; e < 16; ++e) f.u[e] = f2bfu(base[n * stride + kb + e]);
  return f.v;
}

// ---------------- Kernel 1: qkv = x @ qkv_w^T + qkv_b  (bf16 out) ----------------
// M=200704 (=12544 tiles of 16), K=256, N=768. 64x64 block, 8 waves, 2 tiles/wave.
__global__ __launch_bounds__(256) void qkv_gemm_kernel(
    const float* __restrict__ x, const float* __restrict__ w,
    const float* __restrict__ qb, unsigned short* __restrict__ qkv) {
  extern __shared__ char smem[];
  unsigned short* As = (unsigned short*)smem;  // [64][32] bf16
  unsigned short* Bs = As + 64 * 32;           // [64][32] bf16
  const int tid = threadIdx.x, lane = tid & 31, wave = tid >> 5;
  const int R0 = blockIdx.x * 64, C0 = blockIdx.y * 64;
  const int mt = wave & 3, nt0 = (wave >> 2) * 2;
  f32x8 acc[2] = {};
  for (int kk = 0; kk < 8; ++kk) {
    const int k0 = kk * 32;
    // vectorized stage: 4 f32 -> 4 bf16 per iteration (b128 loads, b64 LDS stores)
    for (int i = tid; i < 512; i += 256) {
      int r = i >> 3, k = (i & 7) << 2;
      float4 xa = *(const float4*)&x[(R0 + r) * 256 + k0 + k];
      float4 wb = *(const float4*)&w[(C0 + r) * 256 + k0 + k];
      ushort4 oa = {f2bfu(xa.x), f2bfu(xa.y), f2bfu(xa.z), f2bfu(xa.w)};
      ushort4 ob = {f2bfu(wb.x), f2bfu(wb.y), f2bfu(wb.z), f2bfu(wb.w)};
      *(ushort4*)&As[r * 32 + k] = oa;
      *(ushort4*)&Bs[r * 32 + k] = ob;
    }
    __syncthreads();
    bf16x16 a = load_fragA(As, mt * 16, 32, 0, lane);
#pragma unroll
    for (int t = 0; t < 2; ++t) {
      bf16x16 b = load_fragB_nk(Bs, (nt0 + t) * 16, 32, 0, lane);
      acc[t] = wmma_bf16(a, b, acc[t]);
    }
    __syncthreads();
  }
#pragma unroll
  for (int t = 0; t < 2; ++t) {
    int col = C0 + (nt0 + t) * 16 + (lane & 15);
    float bias = qb[col];
#pragma unroll
    for (int r = 0; r < 8; ++r) {
      int row = R0 + mt * 16 + r + 8 * (lane >> 4);
      qkv[row * 768 + col] = f2bfu(acc[t][r] + bias);
    }
  }
}

// ---------------- Kernel 2: windowed attention, 1 block per window, wave==head ----
__global__ __launch_bounds__(256) void attn_kernel(
    const unsigned short* __restrict__ qkv, const float* __restrict__ mask,
    const float* __restrict__ relb, const float* __restrict__ dw,
    float* __restrict__ out) {
  extern __shared__ char smem[];
  unsigned short* qs = (unsigned short*)smem;  // [8][64][32] bf16 (q pre-scaled)
  unsigned short* ks = qs + 8 * 64 * 32;       // [8][64][32]
  unsigned short* vs = ks + 8 * 64 * 32;       // [8][64][32]
  float* Sb = (float*)(smem + 3 * 8 * 64 * 32 * 2);                       // [8][64][64] f32
  unsigned short* Pb = (unsigned short*)(smem + 3 * 8 * 64 * 32 * 2 + 8 * 64 * 64 * 4);
  const int tid = threadIdx.x, lane = tid & 31, h = tid >> 5;
  const int w = blockIdx.x;
  const int base = w * 49;
  const float qscale = 0.17677669529663689f;  // 32^-0.5

  // Stage q,k,v into per-head [64][32] tiles; 4-wide chunks never straddle a
  // q/k/v or head boundary (all boundaries are multiples of 32 elements).
  for (int i = tid; i < 49 * 192; i += 256) {
    int t = i / 192, c4 = (i - t * 192) << 2;
    ushort4 v = *(const ushort4*)&qkv[(base + t) * 768 + c4];
    int which = c4 >> 8, hh = (c4 & 255) >> 5, d = c4 & 31;
    int idx = (hh * 64 + t) * 32 + d;
    if (which == 0) {
      v.x = f2bfu(bfu2f(v.x) * qscale);
      v.y = f2bfu(bfu2f(v.y) * qscale);
      v.z = f2bfu(bfu2f(v.z) * qscale);
      v.w = f2bfu(bfu2f(v.w) * qscale);
      *(ushort4*)&qs[idx] = v;
    } else if (which == 1) {
      *(ushort4*)&ks[idx] = v;
    } else {
      *(ushort4*)&vs[idx] = v;
    }
  }
  for (int i = tid; i < 8 * 15 * 8; i += 256) {  // zero rows 49..63 (4-wide)
    int hh = i / 120, rem = i - hh * 120;
    int t = 49 + rem / 8, d = (rem & 7) << 2;
    int idx = (hh * 64 + t) * 32 + d;
    ushort4 z = {0, 0, 0, 0};
    *(ushort4*)&qs[idx] = z;
    *(ushort4*)&ks[idx] = z;
    *(ushort4*)&vs[idx] = z;
  }
  __syncthreads();

  const unsigned short* Q = qs + h * 2048;
  const unsigned short* K = ks + h * 2048;
  const unsigned short* V = vs + h * 2048;
  float* S = Sb + h * 4096;
  unsigned short* P = Pb + h * 4096;

  // S = (q*scale) @ k^T : 4x4 tiles of 16x16, K=32 in one WMMA each.
  // K-fragments hoisted: loaded once, reused across all 4 row tiles.
  {
    bf16x16 kb[4];
#pragma unroll
    for (int nt = 0; nt < 4; ++nt) kb[nt] = load_fragB_nk(K, nt * 16, 32, 0, lane);
    for (int mt = 0; mt < 4; ++mt) {
      bf16x16 a = load_fragA(Q, mt * 16, 32, 0, lane);
#pragma unroll
      for (int nt = 0; nt < 4; ++nt) {
        f32x8 c = {};
        c = wmma_bf16(a, kb[nt], c);
        int n = nt * 16 + (lane & 15);
        int mb = mt * 16 + 8 * (lane >> 4);
#pragma unroll
        for (int r = 0; r < 8; ++r) S[(mb + r) * 64 + n] = c[r];
      }
    }
  }
  // softmax(S + rel_bias + window mask) over valid 49 cols; P bf16, cols>=49 zeroed
  const float* mrow0 = mask + (w & 63) * 49 * 49;
  for (int i = lane; i < 49; i += 32) {
    int xi = i / 7, yi = i - xi * 7;
    float mx = -3.0e38f;
    for (int j = 0; j < 49; ++j) {
      int xj = j / 7, yj = j - xj * 7;
      int ridx = (xi - xj + 6) * 13 + (yi - yj + 6);
      float v = S[i * 64 + j] + relb[ridx * 8 + h] + mrow0[i * 49 + j];
      S[i * 64 + j] = v;
      mx = fmaxf(mx, v);
    }
    float sum = 0.f;
    for (int j = 0; j < 49; ++j) {
      float e = expf(S[i * 64 + j] - mx);
      S[i * 64 + j] = e;
      sum += e;
    }
    float inv = 1.0f / sum;
    for (int j = 0; j < 64; ++j)
      P[i * 64 + j] = (j < 49) ? f2bfu(S[i * 64 + j] * inv) : (unsigned short)0;
  }
  // out = P @ V (+ dw residual), K padded to 64 (P cols 49..63 are zero).
  // V-fragments hoisted: 4 fragments loaded once, reused for all row tiles.
  {
    bf16x16 vb[2][2];
#pragma unroll
    for (int kt = 0; kt < 2; ++kt)
#pragma unroll
      for (int nt = 0; nt < 2; ++nt)
        vb[kt][nt] = load_fragB_kn(V, kt * 32, 32, nt * 16, lane);
    for (int mt = 0; mt < 4; ++mt) {
      bf16x16 a0 = load_fragA(P, mt * 16, 64, 0, lane);
      bf16x16 a1 = load_fragA(P, mt * 16, 64, 32, lane);
#pragma unroll
      for (int nt = 0; nt < 2; ++nt) {
        f32x8 c = {};
        c = wmma_bf16(a0, vb[0][nt], c);
        c = wmma_bf16(a1, vb[1][nt], c);
        int n = h * 32 + nt * 16 + (lane & 15);
#pragma unroll
        for (int r = 0; r < 8; ++r) {
          int m = mt * 16 + r + 8 * (lane >> 4);
          if (m < 49) {
            int gi = (base + m) * 256 + n;
            out[gi] = c[r] + dw[gi];
          }
        }
      }
    }
  }
}

// ---------------- Kernel 3: LN -> fc1+GELU -> fc2, out += h (in place) -----------
__global__ __launch_bounds__(256) void mlp_kernel(
    float* __restrict__ out, const float* __restrict__ fc1w,
    const float* __restrict__ fc1b, const float* __restrict__ fc2w,
    const float* __restrict__ fc2b, const float* __restrict__ lng,
    const float* __restrict__ lnb) {
  extern __shared__ char smem[];
  unsigned short* ylds = (unsigned short*)smem;  // [64][256] bf16 LN output
  unsigned short* h1s = ylds + 64 * 256;         // [64][512] bf16 GELU(fc1)
  const int tid = threadIdx.x, lane = tid & 31, wave = tid >> 5;
  const int R0 = blockIdx.x * 64;
  // Warm GL2 with the (shared, 512KB each) weight matrices: global_prefetch_b8
  if (tid < 128) {
    __builtin_prefetch(fc1w + tid * 1024, 0, 1);
    __builtin_prefetch(fc2w + tid * 1024, 0, 1);
  }
  if (tid < 64) {
    const float* xr = out + (R0 + tid) * 256;
    float s = 0.f, s2 = 0.f;
    for (int c = 0; c < 256; c += 4) {
      float4 v = *(const float4*)&xr[c];
      s += v.x + v.y + v.z + v.w;
      s2 += v.x * v.x + v.y * v.y + v.z * v.z + v.w * v.w;
    }
    float mu = s * (1.0f / 256.0f);
    float var = s2 * (1.0f / 256.0f) - mu * mu;
    float rs = rsqrtf(var + 1e-5f);
    for (int c = 0; c < 256; c += 4) {
      float4 v = *(const float4*)&xr[c];
      ushort4 o = {f2bfu((v.x - mu) * rs * lng[c] + lnb[c]),
                   f2bfu((v.y - mu) * rs * lng[c + 1] + lnb[c + 1]),
                   f2bfu((v.z - mu) * rs * lng[c + 2] + lnb[c + 2]),
                   f2bfu((v.w - mu) * rs * lng[c + 3] + lnb[c + 3])};
      *(ushort4*)&ylds[tid * 256 + c] = o;
    }
  }
  __syncthreads();
  {  // h1 = gelu(y @ fc1w^T + fc1b) : [64][512], K=256; A-fragments hoisted
    const int mt = wave & 3, ntb = (wave >> 2) * 16;
    bf16x16 a[8];
#pragma unroll
    for (int kk = 0; kk < 8; ++kk)
      a[kk] = load_fragA(ylds, mt * 16, 256, kk * 32, lane);
    for (int nti = 0; nti < 16; ++nti) {
      const int nt = ntb + nti;
      f32x8 c = {};
#pragma unroll
      for (int kk = 0; kk < 8; ++kk) {
        bf16x16 b = load_fragB_w32(fc1w, nt * 16, 256, kk * 32, lane);
        c = wmma_bf16(a[kk], b, c);
      }
      int n = nt * 16 + (lane & 15);
      float bias = fc1b[n];
#pragma unroll
      for (int r = 0; r < 8; ++r) {
        int m = mt * 16 + r + 8 * (lane >> 4);
        float v = c[r] + bias;
        float g = 0.5f * v * (1.0f + erff(v * 0.70710678118654752f));  // exact gelu
        h1s[m * 512 + n] = f2bfu(g);
      }
    }
  }
  __syncthreads();
  {  // out += h1 @ fc2w^T + fc2b : [64][256], K=512; A-fragments hoisted
    const int mt = wave & 3, ntb = (wave >> 2) * 8;
    bf16x16 a[16];
#pragma unroll
    for (int kk = 0; kk < 16; ++kk)
      a[kk] = load_fragA(h1s, mt * 16, 512, kk * 32, lane);
    for (int nti = 0; nti < 8; ++nti) {
      const int nt = ntb + nti;
      f32x8 c = {};
#pragma unroll
      for (int kk = 0; kk < 16; ++kk) {
        bf16x16 b = load_fragB_w32(fc2w, nt * 16, 512, kk * 32, lane);
        c = wmma_bf16(a[kk], b, c);
      }
      int n = nt * 16 + (lane & 15);
      float bias = fc2b[n];
#pragma unroll
      for (int r = 0; r < 8; ++r) {
        int m = mt * 16 + r + 8 * (lane >> 4);
        int gi = (R0 + m) * 256 + n;
        out[gi] = out[gi] + c[r] + bias;
      }
    }
  }
}

// ---------------- host launch ----------------
extern "C" void kernel_launch(void* const* d_in, const int* in_sizes, int n_in,
                              void* d_out, int out_size, void* d_ws, size_t ws_size,
                              hipStream_t stream) {
  (void)in_sizes; (void)n_in; (void)out_size; (void)ws_size;
  const float* x     = (const float*)d_in[0];
  const float* mask  = (const float*)d_in[1];
  const float* dw    = (const float*)d_in[2];
  const float* qkv_w = (const float*)d_in[3];
  const float* qkv_b = (const float*)d_in[4];
  const float* relb  = (const float*)d_in[5];
  const float* fc1w  = (const float*)d_in[6];
  const float* fc1b  = (const float*)d_in[7];
  const float* fc2w  = (const float*)d_in[8];
  const float* fc2b  = (const float*)d_in[9];
  const float* lng   = (const float*)d_in[10];
  const float* lnb   = (const float*)d_in[11];
  float* out = (float*)d_out;
  unsigned short* qkvbuf = (unsigned short*)d_ws;  // [200704][768] bf16 = 308 MB

  // 1) QKV projection: grid = (200704/64, 768/64)
  qkv_gemm_kernel<<<dim3(3136, 12), 256, 2 * 64 * 32 * 2, stream>>>(
      x, qkv_w, qkv_b, qkvbuf);

  // 2) Window attention: 1 workgroup per window, 8 waves = 8 heads, 288 KB LDS
  size_t attn_smem = (size_t)3 * 8 * 64 * 32 * 2   // q/k/v bf16
                   + (size_t)8 * 64 * 64 * 4       // S f32
                   + (size_t)8 * 64 * 64 * 2;      // P bf16
  attn_kernel<<<4096, 256, attn_smem, stream>>>(qkvbuf, mask, relb, dw, out);

  // 3) LN + MLP + residual, in place on d_out: grid = 200704/64
  mlp_kernel<<<3136, 256, (64 * 256 + 64 * 512) * 2, stream>>>(
      out, fc1w, fc1b, fc2w, fc2b, lng, lnb);
}
